// GAT_ED_58755152609378
// MI455X (gfx1250) — matrix operand order repaired and checked by reading, not compiled, not run on hardware
//
#include <hip/hip_runtime.h>
#include <hip/hip_bf16.h>
#include <math.h>

// ============================================================================
// GAT-ED forward for MI455X (gfx1250, wave32, WMMA).
// All matmul-shaped work uses v_wmma_f32_16x16x32_f16 (f16 A/B, f32 acc).
// Pipeline:
//   0) cvt_weights : f32 -> f16 weight copies (tap-major conv w, split Ww)
//   1) conv+SELU   : per-batch GEMM over 7 taps, writes c[B,N,T] & c^T[B,T,N] f16
//   2) GEMM s_all  : [B*N,256] = c_f16 @ Wcat^T   (GAT score projections)
//   3) GEMM gi_e   : cT @ We_ih^T + be_ih
//   4) GRU scan    : enc -> h_cat[:, :, 0:64]
//   5) GAT attn    : leaky/Wa scores, softmax_j, WMMA attn@V, sigmoid
//                    -> h_cat[:, :, 64:128]
//   6) GEMM gi_m ; 7) GRU scan -> h_enc ; 8) GEMM gi_d ; 9) GRU scan -> h_dec
//  10) GEMM recon -> d_out ; 11) pred MLP -> d_out tail
// ============================================================================

typedef __attribute__((ext_vector_type(16))) _Float16 v16h;
typedef __attribute__((ext_vector_type(8)))  float    v8f;

#define B_  64
#define T_  128
#define N_  64
#define H_  64
#define E_  128
#define K_  7
#define G3_ 192   // 3*H

// ---------------------------------------------------------------- helpers ---
__device__ __forceinline__ int koffA(int p, int hf) {
  // contiguous k order (k = p + 16*hf), matching the B fragments
  return p + 16 * hf;
}

// A fragment from row-major f16 [M x lda], rows m0..m0+15, K chunk kb..kb+31
__device__ __forceinline__ v16h load_a_frag(const _Float16* A, int m0, int lda,
                                            int kb, int lm, int hf) {
  union { v16h v; _Float16 h[16]; } u;
  const _Float16* rowp = A + (size_t)(m0 + lm) * lda + kb;
#pragma unroll
  for (int p = 0; p < 16; ++p) u.h[p] = rowp[koffA(p, hf)];
  return u.v;
}

// B fragment where logical B[k][n] = Bw[n][k] (Bw row-major [Nout x ldb])
__device__ __forceinline__ v16h load_bT_frag(const _Float16* Bw, int n0, int ldb,
                                             int kb, int lm, int hf) {
  union { v16h v; _Float16 h[16]; } u;
  const _Float16* rowp = Bw + (size_t)(n0 + lm) * ldb + kb + 16 * hf;
#pragma unroll
  for (int p = 0; p < 16; ++p) u.h[p] = rowp[p];   // contiguous -> b128 loads
  return u.v;
}

// B fragment from row-major B [K x ldb]: element p -> B[kb+p+16*hf][n0+lm]
__device__ __forceinline__ v16h load_b_frag(const _Float16* Bm, int n0, int ldb,
                                            int kb, int lm, int hf) {
  union { v16h v; _Float16 h[16]; } u;
  const _Float16* p0 = Bm + (size_t)(kb + 16 * hf) * ldb + n0 + lm;
#pragma unroll
  for (int p = 0; p < 16; ++p) u.h[p] = p0[(size_t)p * ldb];
  return u.v;
}

#define WMMA(a, b, c) \
  __builtin_amdgcn_wmma_f32_16x16x32_f16(false, (a), false, (b), (short)0, (c), false, false)

// Branchless SELU: exp(min(x,0))-1 == 0 for x>=0, so no exec-mask divergence.
__device__ __forceinline__ float seluf(float x) {
  const float sc = 1.0507009873554805f, al = 1.6732632423543772f;
  float neg = __expf(fminf(x, 0.f)) - 1.f;
  return sc * (fmaxf(x, 0.f) + al * neg);
}
__device__ __forceinline__ float sigmoidf_(float x) { return 1.f / (1.f + __expf(-x)); }

// ------------------------------------------------------ weight conversion ---
__global__ __launch_bounds__(256) void cvt_weights_kernel(
    const float* __restrict__ We_ih, const float* __restrict__ We_hh,
    const float* __restrict__ Wm_ih, const float* __restrict__ Wm_hh,
    const float* __restrict__ Wd_ih, const float* __restrict__ Wd_hh,
    const float* __restrict__ Ww,    const float* __restrict__ W_dec,
    const float* __restrict__ conv_w,
    _Float16* hWe_ih, _Float16* hWe_hh, _Float16* hWm_ih, _Float16* hWm_hh,
    _Float16* hWd_ih, _Float16* hWd_hh, _Float16* hWcat,  _Float16* hWdec,
    _Float16* cw_taps) {
  int which = blockIdx.x, tid = threadIdx.x;
  switch (which) {
    case 0: for (int i = tid; i < G3_*N_;  i += 256) hWe_ih[i] = (_Float16)We_ih[i]; break;
    case 1: for (int i = tid; i < G3_*H_;  i += 256) hWe_hh[i] = (_Float16)We_hh[i]; break;
    case 2: for (int i = tid; i < G3_*128; i += 256) hWm_ih[i] = (_Float16)Wm_ih[i]; break;
    case 3: for (int i = tid; i < G3_*H_;  i += 256) hWm_hh[i] = (_Float16)Wm_hh[i]; break;
    case 4: for (int i = tid; i < G3_*H_;  i += 256) hWd_ih[i] = (_Float16)Wd_ih[i]; break;
    case 5: for (int i = tid; i < G3_*H_;  i += 256) hWd_hh[i] = (_Float16)Wd_hh[i]; break;
    case 6: // Wcat[e][t]: rows 0..127 = Ww[:, :T], rows 128..255 = Ww[:, T:]
      for (int i = tid; i < 256 * T_; i += 256) {
        int e = i >> 7, t = i & 127;
        hWcat[i] = (_Float16)Ww[(size_t)(e & 127) * (2 * T_) + (e >> 7) * T_ + t];
      }
      break;
    case 7: for (int i = tid; i < N_*H_; i += 256) hWdec[i] = (_Float16)W_dec[i]; break;
    case 8: // cw_taps[kk][o][in] from conv_w[o][in][kk]
      for (int i = tid; i < K_ * N_ * N_; i += 256) {
        int kk = i / (N_ * N_), rem = i % (N_ * N_), o = rem >> 6, in = rem & 63;
        cw_taps[i] = (_Float16)conv_w[((size_t)o * N_ + in) * K_ + kk];
      }
      break;
    default: break;
  }
}

// ----------------------------------------------------------- conv + SELU ----
// grid = B, block = 256 (8 waves). c[b,n,t] = selu(sum_k W_k @ x_shifted + b)
__global__ __launch_bounds__(256) void conv_selu_kernel(
    const float* __restrict__ x,            // [B,T,N] f32
    const _Float16* __restrict__ cw_taps,   // [7][64][64] f16
    const float* __restrict__ conv_b,       // [N]
    _Float16* __restrict__ c_f16,           // [B,N,T]
    _Float16* __restrict__ cT_f16) {        // [B,T,N]
  __shared__ _Float16 xT[N_][T_ + 8];       // padded: col c holds t = c-3
  int b = blockIdx.x, tid = threadIdx.x;
  // warm L2/WGP$ for the tap weights (reused by every block)
  __builtin_prefetch(cw_taps + (size_t)(tid & 7) * N_ * N_, 0, 3);
  for (int i = tid; i < N_ * (T_ + 8); i += 256) {
    int n = i / (T_ + 8), col = i % (T_ + 8);
    int t = col - 3;
    float v = (t >= 0 && t < T_) ? x[(size_t)b * T_ * N_ + (size_t)t * N_ + n] : 0.f;
    xT[n][col] = (_Float16)v;
  }
  __syncthreads();
  int wave = tid >> 5, lane = tid & 31, lm = lane & 15, hf = lane >> 4;
  for (int q = 0; q < 4; ++q) {
    int tile = wave * 4 + q;
    int m0 = (tile >> 3) * 16;   // n_out tile
    int t0 = (tile & 7) * 16;    // time tile
    v8f acc = {};
    for (int kk = 0; kk < K_; ++kk) {
      const _Float16* Aw = cw_taps + (size_t)kk * N_ * N_;
      for (int kb = 0; kb < N_; kb += 32) {
        v16h a = load_a_frag(Aw, m0, N_, kb, lm, hf);
        v16h bf = load_b_frag(&xT[0][0], t0 + kk, T_ + 8, kb, lm, hf);
        acc = WMMA(a, bf, acc);
      }
    }
#pragma unroll
    for (int r = 0; r < 8; ++r) {
      int n_out = m0 + r + 8 * hf;
      int t = t0 + lm;
      _Float16 hv = (_Float16)seluf(acc[r] + conv_b[n_out]);
      c_f16 [(size_t)b * N_ * T_ + (size_t)n_out * T_ + t]    = hv;
      cT_f16[(size_t)b * T_ * N_ + (size_t)t * N_ + n_out]    = hv;
    }
  }
}

// ----------------------------------------------------------- generic GEMM ---
// C[M,Nout] = A[M,K](f16) @ Bw[Nout,K]^T(f16) (+ bias), f32 out.
// grid = (Nout/64, M/64), block = 256 (8 waves, 2 tiles each).
__global__ __launch_bounds__(256) void gemm_f16_kernel(
    const _Float16* __restrict__ A, int lda,
    const _Float16* __restrict__ Bw, int ldb,
    const float* __restrict__ bias,
    float* __restrict__ C, int ldc, int K) {
  int bx = blockIdx.x, by = blockIdx.y;
  int tid = threadIdx.x, wave = tid >> 5, lane = tid & 31, lm = lane & 15, hf = lane >> 4;
  // prefetch this block's operand rows (global_prefetch_b8)
  __builtin_prefetch(A + (size_t)(by * 64 + (tid & 63)) * lda, 0, 3);
  __builtin_prefetch(Bw + (size_t)(bx * 64 + (tid & 63)) * ldb, 0, 3);
  for (int q = 0; q < 2; ++q) {
    int tile = wave * 2 + q;                 // 0..15 in 4x4 grid
    int m0 = by * 64 + (tile >> 2) * 16;
    int n0 = bx * 64 + (tile & 3) * 16;
    v8f acc = {};
    for (int kb = 0; kb < K; kb += 32) {
      v16h a = load_a_frag(A, m0, lda, kb, lm, hf);
      v16h bf = load_bT_frag(Bw, n0, ldb, kb, lm, hf);
      acc = WMMA(a, bf, acc);
    }
    float bn = bias ? bias[n0 + lm] : 0.f;
#pragma unroll
    for (int r = 0; r < 8; ++r) {
      int m = m0 + r + 8 * hf;
      C[(size_t)m * ldc + n0 + lm] = acc[r] + bn;
    }
  }
}

// ------------------------------------------------------------- GRU scan -----
// grid = 4 (16 batches each), block = 128 (4 waves). h resident in LDS.
__global__ __launch_bounds__(128) void gru_scan_kernel(
    const float* __restrict__ gi,          // [B,T,192] precomputed input proj
    const _Float16* __restrict__ Whh,      // [192,64] f16
    const float* __restrict__ bhh,         // [192]
    _Float16* __restrict__ out, int ldo, int out_off) {
  __shared__ float    h32[16][H_];
  __shared__ _Float16 h16[16][H_];
  __shared__ float    gh [16][G3_];
  int bb = blockIdx.x, tid = threadIdx.x;
  int wave = tid >> 5, lane = tid & 31, lm = lane & 15, hf = lane >> 4;
  for (int i = tid; i < 16 * H_; i += 128) {
    h32[i >> 6][i & 63] = 0.f;  h16[i >> 6][i & 63] = (_Float16)0.f;
  }
  __syncthreads();
  for (int t = 0; t < T_; ++t) {
    // gh = h @ Whh^T : 12 col tiles of 16, 3 per wave, K=64 -> 2 WMMA each
    for (int j = 0; j < 3; ++j) {
      int n0 = (wave * 3 + j) * 16;
      v8f acc = {};
      for (int kb = 0; kb < H_; kb += 32) {
        v16h a = load_a_frag(&h16[0][0], 0, H_, kb, lm, hf);
        v16h bf = load_bT_frag(Whh, n0, H_, kb, lm, hf);
        acc = WMMA(a, bf, acc);
      }
#pragma unroll
      for (int r = 0; r < 8; ++r) gh[r + 8 * hf][n0 + lm] = acc[r];
    }
    __syncthreads();
    // fused gate math: 16*64 elements, 8 per thread
    for (int i = tid; i < 16 * H_; i += 128) {
      int bl = i >> 6, hc = i & 63;
      int bg = bb * 16 + bl;
      const float* g = gi + ((size_t)bg * T_ + t) * G3_;
      float g_r = gh[bl][hc]        + bhh[hc];
      float g_z = gh[bl][H_ + hc]   + bhh[H_ + hc];
      float g_n = gh[bl][2*H_ + hc] + bhh[2*H_ + hc];
      float r = sigmoidf_(g[hc] + g_r);
      float z = sigmoidf_(g[H_ + hc] + g_z);
      float n = tanhf(g[2*H_ + hc] + r * g_n);
      float hn = (1.f - z) * n + z * h32[bl][hc];
      h32[bl][hc] = hn;
      h16[bl][hc] = (_Float16)hn;
      out[((size_t)bg * T_ + t) * ldo + out_off + hc] = (_Float16)hn;
    }
    __syncthreads();
  }
}

// ------------------------------------------------------------- GAT attn -----
// grid = B, block = 256. e[i,j] = Wa.leaky(s_i[i]+s_j[j]+bw)+ba ; softmax_j ;
// h_mp = sigmoid(attn @ v) via WMMA ; write h_cat[:, :, 64:128].
__global__ __launch_bounds__(256) void gat_attn_kernel(
    const float* __restrict__ s_all,       // [B*N, 256] (cols 0:128 s_i, 128:256 s_j)
    const float* __restrict__ bw, const float* __restrict__ Wa,
    const float* __restrict__ ba,
    const _Float16* __restrict__ c_f16,    // [B,N,T]
    _Float16* __restrict__ h_cat) {        // [B,T,128]
  __shared__ _Float16 si[N_][E_];
  __shared__ _Float16 sj[N_][E_];
  __shared__ float    e_att[N_][N_];
  __shared__ _Float16 attn16[N_][N_];
  __shared__ float    bw_s[E_], wa_s[E_];
  int b = blockIdx.x, tid = threadIdx.x;
  for (int i = tid; i < N_ * E_; i += 256) {
    int node = i >> 7, e = i & 127;
    const float* sp = s_all + ((size_t)b * N_ + node) * 256;
    si[node][e] = (_Float16)sp[e];
    sj[node][e] = (_Float16)sp[E_ + e];
  }
  if (tid < E_) { bw_s[tid] = bw[tid]; wa_s[tid] = Wa[tid]; }
  __syncthreads();
  float ba0 = ba[0];
  for (int idx = tid; idx < N_ * N_; idx += 256) {
    int i = idx >> 6, j = idx & 63;
    float acc = ba0;
    for (int e = 0; e < E_; ++e) {
      float tv = (float)si[i][e] + (float)sj[j][e] + bw_s[e];
      acc += wa_s[e] * (tv > 0.f ? tv : 0.2f * tv);   // cndmask, no divergence
    }
    e_att[i][j] = acc;
  }
  __syncthreads();
  if (tid < N_) {                            // softmax over j for row tid
    int i = tid;
    float mx = -3.4e38f;
    for (int j = 0; j < N_; ++j) mx = fmaxf(mx, e_att[i][j]);
    float s = 0.f;
    for (int j = 0; j < N_; ++j) s += __expf(e_att[i][j] - mx);
    float inv = 1.f / s;
    for (int j = 0; j < N_; ++j) attn16[i][j] = (_Float16)(__expf(e_att[i][j] - mx) * inv);
  }
  __syncthreads();
  int wave = tid >> 5, lane = tid & 31, lm = lane & 15, hf = lane >> 4;
  for (int q = 0; q < 4; ++q) {
    int tile = wave * 4 + q;
    int i0 = (tile >> 3) * 16;
    int t0 = (tile & 7) * 16;
    v8f acc = {};
    for (int kb = 0; kb < N_; kb += 32) {
      v16h a = load_a_frag(&attn16[0][0], i0, N_, kb, lm, hf);
      v16h bf = load_b_frag(c_f16 + (size_t)b * N_ * T_, t0, T_, kb, lm, hf);
      acc = WMMA(a, bf, acc);
    }
#pragma unroll
    for (int r = 0; r < 8; ++r) {
      int nodei = i0 + r + 8 * hf;
      int t = t0 + lm;
      h_cat[((size_t)b * T_ + t) * 128 + H_ + nodei] = (_Float16)sigmoidf_(acc[r]);
    }
  }
}

// ------------------------------------------------------------- pred MLP -----
__global__ __launch_bounds__(256) void pred_kernel(
    const _Float16* __restrict__ h_dec16,  // [B,T,H]
    const float* __restrict__ W_p1, const float* __restrict__ b_p1,
    const float* __restrict__ W_p2, const float* __restrict__ b_p2,
    float* __restrict__ out_pred) {        // [B,N]
  __shared__ float p[B_][N_];
  int tid = threadIdx.x;
  for (int idx = tid; idx < B_ * N_; idx += 256) {
    int b = idx >> 6, n = idx & 63;
    const _Float16* h = h_dec16 + (size_t)b * T_ * H_;   // t = 0
    float acc = b_p1[n];
    for (int k = 0; k < H_; ++k) acc += (float)h[k] * W_p1[(size_t)n * H_ + k];
    p[b][n] = fmaxf(acc, 0.f);
  }
  __syncthreads();
  for (int idx = tid; idx < B_ * N_; idx += 256) {
    int b = idx >> 6, n = idx & 63;
    float acc = b_p2[n];
    for (int m = 0; m < N_; ++m) acc += p[b][m] * W_p2[(size_t)n * N_ + m];
    out_pred[(size_t)b * N_ + n] = acc;
  }
}

// ------------------------------------------------------------- launcher -----
extern "C" void kernel_launch(void* const* d_in, const int* in_sizes, int n_in,
                              void* d_out, int out_size, void* d_ws, size_t ws_size,
                              hipStream_t stream) {
  (void)in_sizes; (void)n_in; (void)out_size; (void)ws_size;
  const float* x      = (const float*)d_in[0];
  const float* conv_w = (const float*)d_in[2];
  const float* conv_b = (const float*)d_in[3];
  const float* We_ih  = (const float*)d_in[4];
  const float* We_hh  = (const float*)d_in[5];
  const float* be_ih  = (const float*)d_in[6];
  const float* be_hh  = (const float*)d_in[7];
  const float* Ww     = (const float*)d_in[8];
  const float* bw     = (const float*)d_in[9];
  const float* Wa     = (const float*)d_in[10];
  const float* ba     = (const float*)d_in[11];
  const float* Wm_ih  = (const float*)d_in[12];
  const float* Wm_hh  = (const float*)d_in[13];
  const float* bm_ih  = (const float*)d_in[14];
  const float* bm_hh  = (const float*)d_in[15];
  const float* Wd_ih  = (const float*)d_in[16];
  const float* Wd_hh  = (const float*)d_in[17];
  const float* bd_ih  = (const float*)d_in[18];
  const float* bd_hh  = (const float*)d_in[19];
  const float* b_dec  = (const float*)d_in[21];
  const float* W_p1   = (const float*)d_in[22];
  const float* b_p1   = (const float*)d_in[23];
  const float* W_p2   = (const float*)d_in[24];
  const float* b_p2   = (const float*)d_in[25];

  // ---- workspace carve-up (deterministic) ----
  char* ws = (char*)d_ws;
  size_t off = 0;
  auto take = [&](size_t bytes) -> char* {
    char* p = ws + off;
    off = (off + bytes + 255) & ~(size_t)255;
    return p;
  };
  _Float16* hWe_ih = (_Float16*)take((size_t)G3_ * N_ * 2);
  _Float16* hWe_hh = (_Float16*)take((size_t)G3_ * H_ * 2);
  _Float16* hWm_ih = (_Float16*)take((size_t)G3_ * 128 * 2);
  _Float16* hWm_hh = (_Float16*)take((size_t)G3_ * H_ * 2);
  _Float16* hWd_ih = (_Float16*)take((size_t)G3_ * H_ * 2);
  _Float16* hWd_hh = (_Float16*)take((size_t)G3_ * H_ * 2);
  _Float16* hWcat  = (_Float16*)take((size_t)256 * T_ * 2);
  _Float16* hWdec  = (_Float16*)take((size_t)N_ * H_ * 2);
  _Float16* cwtaps = (_Float16*)take((size_t)K_ * N_ * N_ * 2);
  _Float16* c_f16  = (_Float16*)take((size_t)B_ * N_ * T_ * 2);
  _Float16* cT_f16 = (_Float16*)take((size_t)B_ * T_ * N_ * 2);
  float*    f_s    = (float*)   take((size_t)B_ * N_ * 256 * 4);
  float*    f_gi   = (float*)   take((size_t)B_ * T_ * G3_ * 4);
  _Float16* h_cat  = (_Float16*)take((size_t)B_ * T_ * 128 * 2);
  _Float16* h_enc  = (_Float16*)take((size_t)B_ * T_ * H_ * 2);
  _Float16* h_dec  = (_Float16*)take((size_t)B_ * T_ * H_ * 2);

  // 0) weight conversion
  cvt_weights_kernel<<<9, 256, 0, stream>>>(
      We_ih, We_hh, Wm_ih, Wm_hh, Wd_ih, Wd_hh, Ww, (const float*)d_in[20], conv_w,
      hWe_ih, hWe_hh, hWm_ih, hWm_hh, hWd_ih, hWd_hh, hWcat, hWdec, cwtaps);

  // 1) conv + SELU
  conv_selu_kernel<<<B_, 256, 0, stream>>>(x, cwtaps, conv_b, c_f16, cT_f16);

  // 2) GAT score projection: s_all[B*N,256] = c @ Wcat^T
  gemm_f16_kernel<<<dim3(4, 64), 256, 0, stream>>>(c_f16, T_, hWcat, T_,
                                                   nullptr, f_s, 256, T_);
  // 3) enc GRU input proj
  gemm_f16_kernel<<<dim3(3, 128), 256, 0, stream>>>(cT_f16, N_, hWe_ih, N_,
                                                    be_ih, f_gi, G3_, N_);
  // 4) enc GRU scan -> h_cat[:, :, 0:64]
  gru_scan_kernel<<<4, 128, 0, stream>>>(f_gi, hWe_hh, be_hh, h_cat, 128, 0);

  // 5) GAT attention -> h_cat[:, :, 64:128]
  gat_attn_kernel<<<B_, 256, 0, stream>>>(f_s, bw, Wa, ba, c_f16, h_cat);

  // 6) mid GRU input proj (K=128)
  gemm_f16_kernel<<<dim3(3, 128), 256, 0, stream>>>(h_cat, 128, hWm_ih, 128,
                                                    bm_ih, f_gi, G3_, 128);
  // 7) mid GRU scan -> h_enc
  gru_scan_kernel<<<4, 128, 0, stream>>>(f_gi, hWm_hh, bm_hh, h_enc, H_, 0);

  // 8) dec GRU input proj
  gemm_f16_kernel<<<dim3(3, 128), 256, 0, stream>>>(h_enc, H_, hWd_ih, H_,
                                                    bd_ih, f_gi, G3_, H_);
  // 9) dec GRU scan -> h_dec
  gru_scan_kernel<<<4, 128, 0, stream>>>(f_gi, hWd_hh, bd_hh, h_dec, H_, 0);

  // 10) recon = h_dec @ W_dec^T + b_dec  -> d_out[0 : B*T*N]
  gemm_f16_kernel<<<dim3(1, 128), 256, 0, stream>>>(h_dec, H_, hWdec, H_,
                                                    b_dec, (float*)d_out, N_, H_);
  // 11) pred MLP -> d_out[B*T*N : B*T*N + B*N]
  pred_kernel<<<1, 256, 0, stream>>>(h_dec, W_p1, b_p1, W_p2, b_p2,
                                     (float*)d_out + (size_t)B_ * T_ * N_);
}
